// GMMINR_21457656611259
// MI455X (gfx1250) — matrix operand order repaired and checked by reading, not compiled, not run on hardware
//
#include <hip/hip_runtime.h>
#include <hip/hip_bf16.h>

// MI455X / gfx1250, wave32. Fully fused GMM-density -> feats GEMM -> 4x SIREN
// sine layers -> tanh head. All GEMMs on v_wmma_f32_16x16x32_f16; activations
// stay in LDS (ping-pong 64x512 f16 tiles); weights pre-packed to f16 WMMA
// B-fragment layout once per launch (L2-resident thereafter).
// Each wave accumulates 4 N-tiles per A-fragment load: 4 WMMA per ds_load pair,
// independent accumulators hide WMMA->WMMA RAW hazards.

typedef __attribute__((ext_vector_type(16))) _Float16 v16h;
typedef __attribute__((ext_vector_type(8)))  _Float16 v8h;
typedef __attribute__((ext_vector_type(8)))  float    v8f;

#define OMEGA 30.0f
#define ROWS_PER_BLOCK 64
#define THREADS 256

// ---------------------------------------------------------------------------
// Prep: pack a row-major f32 weight [Ksrc x Ncols] into f16 WMMA B-fragments.
// Packed layout: [KT][NT][lane 0..31][16 halfs], where for lane l, half i:
//   k = kk*32 + (l<16 ? 0 : 16) + i ,  n = nt*16 + (l&15)
// (B 32x16 in this code's contiguous K order, same as A: lane = column, 16 contiguous K values per lane.)
// Rows k >= Ksrc are zero-padded (layer-0 K: 131 -> 160).
// ---------------------------------------------------------------------------
__global__ void pack_weights_f16(const float* __restrict__ src,
                                 _Float16* __restrict__ dst,
                                 int Ksrc, int KT, int NT, int Ncols)
{
    int total = KT * NT * 32 * 16;
    for (int idx = blockIdx.x * blockDim.x + threadIdx.x; idx < total;
         idx += gridDim.x * blockDim.x) {
        int i  = idx & 15;
        int l  = (idx >> 4) & 31;
        int t  = idx >> 9;
        int nt = t % NT;
        int kk = t / NT;
        int k  = kk * 32 + ((l < 16) ? 0 : 16) + i;
        int n  = nt * 16 + (l & 15);
        float v = (k < Ksrc) ? src[k * Ncols + n] : 0.0f;
        dst[idx] = (_Float16)v;
    }
}

// coeff[g] = sqrt(det(P_g)) / (2*pi)^(3/2)
__global__ void gauss_coeff_kernel(const float* __restrict__ P,
                                   float* __restrict__ coeff, int G)
{
    int g = blockIdx.x * blockDim.x + threadIdx.x;
    if (g >= G) return;
    const float* p = P + g * 9;
    float det = p[0] * (p[4] * p[8] - p[5] * p[7])
              - p[1] * (p[3] * p[8] - p[5] * p[6])
              + p[2] * (p[3] * p[7] - p[4] * p[6]);
    const float inv_2pi_3_2 = 0.06349363593424097f; // (2*pi)^(-1.5)
    coeff[g] = sqrtf(det) * inv_2pi_3_2;
}

// ---------------------------------------------------------------------------
// Workgroup GEMM: OUT[64 x NT*16] = act( IN[64 x KT*32] @ Wp + bias )
// IN/OUT in LDS (row-major, strides in halfs). Wp = packed f16 B-fragments.
// 8 waves cover 4 M-tiles x NT N-tiles; each wave iteration owns one M-tile
// and FOUR consecutive N-tiles, so each A fragment (2x ds_load_b128) feeds
// 4 independent v_wmma accumulations.
// mode 0: out = scale * acc          (feats layer)
// mode 1: out = sin(OMEGA*(acc+b))   (SIREN layers)
// NT must be a multiple of 4.
// ---------------------------------------------------------------------------
__device__ __forceinline__ void block_gemm(
    const _Float16* __restrict__ inL, int inStride,
    _Float16* __restrict__ outL, int outStride,
    const _Float16* __restrict__ Wp, const float* __restrict__ bias,
    int KT, int NT, int mode, float scale)
{
    const int tid     = threadIdx.x;
    const int lane    = tid & 31;
    const int wave    = tid >> 5;
    const int lanem   = lane & 15;
    const int halfSel = (lane < 16) ? 0 : 16;  // A-fragment K sub-select
    const int nGrp    = NT >> 2;               // N-tile groups of 4
    const int nWork   = 4 * nGrp;              // (4 M-tiles) x groups

    for (int t = wave; t < nWork; t += 8) {
        const int mt = t / nGrp;
        const int ng = t % nGrp;               // tiles nt = ng*4 + j
        v8f acc0 = {}, acc1 = {}, acc2 = {}, acc3 = {};

        const _Float16* arow = inL + (mt * 16 + lanem) * inStride;
        const _Float16* bgrp = Wp + ((size_t)ng * 4 * 32 * 16) + lane * 16;
        const size_t kstride = (size_t)NT * 32 * 16;

        for (int kk = 0; kk < KT; ++kk) {
            const int kb = kk * 32;
            // A fragment (16x32 f16): two ds_load_b128 per lane
            v8h a0 = *(const v8h*)(arow + kb + halfSel);
            v8h a1 = *(const v8h*)(arow + kb + halfSel + 8);
            v16h a = __builtin_shufflevector(a0, a1,
                       0,1,2,3,4,5,6,7,8,9,10,11,12,13,14,15);
            // 4 B fragments: 32 contiguous bytes per lane each
            const _Float16* bk = bgrp + kk * kstride;
            v16h b0 = *(const v16h*)(bk);
            v16h b1 = *(const v16h*)(bk + 32 * 16);
            v16h b2 = *(const v16h*)(bk + 2 * 32 * 16);
            v16h b3 = *(const v16h*)(bk + 3 * 32 * 16);
            acc0 = __builtin_amdgcn_wmma_f32_16x16x32_f16(
                       false, a, false, b0, (short)0, acc0, false, false);
            acc1 = __builtin_amdgcn_wmma_f32_16x16x32_f16(
                       false, a, false, b1, (short)0, acc1, false, false);
            acc2 = __builtin_amdgcn_wmma_f32_16x16x32_f16(
                       false, a, false, b2, (short)0, acc2, false, false);
            acc3 = __builtin_amdgcn_wmma_f32_16x16x32_f16(
                       false, a, false, b3, (short)0, acc3, false, false);
        }

        // D layout: VGPR r -> row (mt*16 + r + (lane<16?0:8)), col nt*16+(lane&15)
        const int rbase = mt * 16 + ((lane < 16) ? 0 : 8);
#pragma unroll
        for (int j = 0; j < 4; ++j) {
            v8f acc = (j == 0) ? acc0 : (j == 1) ? acc1 : (j == 2) ? acc2 : acc3;
            const int col  = (ng * 4 + j) * 16 + lanem;
            const float bz = (mode == 1) ? bias[col] : 0.0f;
#pragma unroll
            for (int r = 0; r < 8; ++r) {
                float v = acc[r];
                v = (mode == 1) ? __sinf(OMEGA * (v + bz)) : v * scale;
                outL[(rbase + r) * outStride + col] = (_Float16)v;
            }
        }
    }
}

// ---------------------------------------------------------------------------
// Fused kernel: one block of 64 rows end-to-end.
// ---------------------------------------------------------------------------
__launch_bounds__(THREADS)
__global__ void gmm_siren_fused(
    const float* __restrict__ x,         // [N,3]
    const float* __restrict__ centers,   // [256,3]
    const float* __restrict__ precision, // [256,9]
    const float* __restrict__ coeff,     // [256]
    const _Float16* __restrict__ Fp,     // packed features [8][8][32][16]
    const _Float16* __restrict__ W0p,    // [5][32][32][16]  (K padded 131->160)
    const float* __restrict__ b0,
    const _Float16* __restrict__ W1p, const float* __restrict__ b1,
    const _Float16* __restrict__ W2p, const float* __restrict__ b2,
    const _Float16* __restrict__ W3p, const float* __restrict__ b3,
    const float* __restrict__ Wout,      // [512,3]
    const float* __restrict__ bout,      // [3]
    float* __restrict__ y,               // [N,3]
    int nrows)
{
    extern __shared__ char smem[];
    // LDS carve (bytes): gp 13312 | xr 768 | bufA 65536 | bufB 65536 = 145152
    float*     gp   = (float*)smem;                    // [256][13]; reused for Wout
    float*     xr   = (float*)(smem + 13312);          // [64][3]
    _Float16*  bufA = (_Float16*)(smem + 14080);       // [64][<=512]
    _Float16*  bufB = (_Float16*)(smem + 79616);       // [64][512]; first holds dens
    float*     WoutL = gp;                             // [512*3] + bout[3] (reuse)

    const int tid  = threadIdx.x;
    const int row0 = blockIdx.x * ROWS_PER_BLOCK;

    // Stage Gaussian params: thread t owns gaussian t (G == THREADS == 256)
    {
        const float* c = centers + tid * 3;
        const float* p = precision + tid * 9;
        float* d = gp + tid * 13;
        d[0] = c[0]; d[1] = c[1]; d[2] = c[2];
#pragma unroll
        for (int j = 0; j < 9; ++j) d[3 + j] = p[j];
        d[12] = coeff[tid];
    }
    // Stage 64 rows of x
    if (tid < ROWS_PER_BLOCK * 3) {
        int gi = row0 * 3 + tid;
        xr[tid] = (gi < nrows * 3) ? x[gi] : 0.0f;
    }
    __syncthreads();

    // Density tile: thread t = gaussian g for all 64 rows -> bufB[64][256] f16
    {
        const float* d = gp + tid * 13;
        float c0 = d[0], c1 = d[1], c2 = d[2];
        float p00 = d[3], p01 = d[4], p02 = d[5];
        float p10 = d[6], p11 = d[7], p12 = d[8];
        float p20 = d[9], p21 = d[10], p22 = d[11];
        float cf = d[12];
        for (int n = 0; n < ROWS_PER_BLOCK; ++n) {
            float e0 = xr[n * 3 + 0] - c0;
            float e1 = xr[n * 3 + 1] - c1;
            float e2 = xr[n * 3 + 2] - c2;
            float q = e0 * (p00 * e0 + p01 * e1 + p02 * e2)
                    + e1 * (p10 * e0 + p11 * e1 + p12 * e2)
                    + e2 * (p20 * e0 + p21 * e1 + p22 * e2);
            bufB[n * 256 + tid] = (_Float16)(cf * __expf(-0.5f * q));
        }
    }
    __syncthreads();

    // feats = (dens @ features) * sqrt(6/256) -> bufA cols [0,128), stride 160
    block_gemm(bufB, 256, bufA, 160, Fp, nullptr, 8, 8, 0, 0.15309310892f);
    // cols 128..130 = x ; cols 131..159 = 0 (matches zero-padded W0 rows)
    for (int e = tid; e < ROWS_PER_BLOCK * 32; e += THREADS) {
        int rr = e >> 5;
        int cc = 128 + (e & 31);
        float v = (cc < 131) ? xr[rr * 3 + (cc - 128)] : 0.0f;
        bufA[rr * 160 + cc] = (_Float16)v;
    }
    // gp is dead now -> stage output head into LDS
    for (int e = tid; e < 512 * 3 + 3; e += THREADS)
        WoutL[e] = (e < 1536) ? Wout[e] : bout[e - 1536];
    __syncthreads();

    // SIREN layers (ping-pong LDS buffers)
    block_gemm(bufA, 160, bufB, 512, W0p, b0, 5, 32, 1, 0.0f);
    __syncthreads();
    block_gemm(bufB, 512, bufA, 512, W1p, b1, 16, 32, 1, 0.0f);
    __syncthreads();
    block_gemm(bufA, 512, bufB, 512, W2p, b2, 16, 32, 1, 0.0f);
    __syncthreads();
    block_gemm(bufB, 512, bufA, 512, W3p, b3, 16, 32, 1, 0.0f);
    __syncthreads();

    // Head: y[r,o] = tanh( sum_k h[r,k] * Wout[k,o] + bout[o] )
    if (tid < ROWS_PER_BLOCK * 3) {
        int rr = tid / 3, o = tid % 3;
        const v8h* hrow = (const v8h*)(bufA + rr * 512);
        float s = WoutL[1536 + o];
        for (int c = 0; c < 64; ++c) {
            v8h hv = hrow[c];
#pragma unroll
            for (int i = 0; i < 8; ++i)
                s += (float)hv[i] * WoutL[(c * 8 + i) * 3 + o];
        }
        int gr = row0 + rr;
        if (gr < nrows) y[gr * 3 + o] = tanhf(s);
    }
}

// ---------------------------------------------------------------------------
extern "C" void kernel_launch(void* const* d_in, const int* in_sizes, int n_in,
                              void* d_out, int out_size, void* d_ws, size_t ws_size,
                              hipStream_t stream)
{
    const float* x         = (const float*)d_in[0];
    const float* centers   = (const float*)d_in[1];
    const float* precision = (const float*)d_in[2];
    const float* features  = (const float*)d_in[3];
    const float* W0        = (const float*)d_in[4];
    const float* b0        = (const float*)d_in[5];
    const float* W1        = (const float*)d_in[6];
    const float* b1        = (const float*)d_in[7];
    const float* W2        = (const float*)d_in[8];
    const float* b2        = (const float*)d_in[9];
    const float* W3        = (const float*)d_in[10];
    const float* b3        = (const float*)d_in[11];
    const float* Wout      = (const float*)d_in[12];
    const float* bout      = (const float*)d_in[13];
    float* y = (float*)d_out;

    const int nrows = in_sizes[0] / 3;

    // Workspace carve (all f16 packed weights + coeff): ~1.8 MB
    char* ws = (char*)d_ws;
    _Float16* Fp  = (_Float16*)ws; ws += (size_t)8  * 8  * 32 * 16 * 2; //  64 KB
    _Float16* W0p = (_Float16*)ws; ws += (size_t)5  * 32 * 32 * 16 * 2; // 160 KB
    _Float16* W1p = (_Float16*)ws; ws += (size_t)16 * 32 * 32 * 16 * 2; // 512 KB
    _Float16* W2p = (_Float16*)ws; ws += (size_t)16 * 32 * 32 * 16 * 2;
    _Float16* W3p = (_Float16*)ws; ws += (size_t)16 * 32 * 32 * 16 * 2;
    float*   coeff = (float*)ws;   ws += 256 * 4;

    pack_weights_f16<<<64,  THREADS, 0, stream>>>(features, Fp,  256, 8,  8,  128);
    pack_weights_f16<<<64,  THREADS, 0, stream>>>(W0,       W0p, 131, 5,  32, 512);
    pack_weights_f16<<<128, THREADS, 0, stream>>>(W1,       W1p, 512, 16, 32, 512);
    pack_weights_f16<<<128, THREADS, 0, stream>>>(W2,       W2p, 512, 16, 32, 512);
    pack_weights_f16<<<128, THREADS, 0, stream>>>(W3,       W3p, 512, 16, 32, 512);
    gauss_coeff_kernel<<<1, 256, 0, stream>>>(precision, coeff, 256);

    const int    grid  = (nrows + ROWS_PER_BLOCK - 1) / ROWS_PER_BLOCK; // 2048
    const size_t lds   = 145152; // gp 13312 + xr 768 + bufA 65536 + bufB 65536
    gmm_siren_fused<<<grid, THREADS, lds, stream>>>(
        x, centers, precision, coeff, Fp, W0p, b0, W1p, b1, W2p, b2, W3p, b3,
        Wout, bout, y, nrows);
}
